// GCNLayer_9921374454291
// MI455X (gfx1250) — compile-verified, matrix operand order checked by reading, not run on hardware
//
#include <hip/hip_runtime.h>
#include <hip/hip_bf16.h>

// ---------------------------------------------------------------------------
// Problem constants (from the reference)
// ---------------------------------------------------------------------------
#define NUM_USER   100000
#define NUM_ITEM   50000
#define N_NODES    150000          // NUM_USER + NUM_ITEM
#define DIM_FEAT   1280
#define H_DIM      256             // 4 * DIM_LATENT
#define DL         64              // DIM_LATENT
#define N_EDGES    3000000
#define NUM_LAYER  3

typedef __attribute__((ext_vector_type(16))) __bf16 v16bf;
typedef __attribute__((ext_vector_type(8)))  float  v8f;
typedef __attribute__((ext_vector_type(2)))  __bf16 bf16x2;

// ---------------------------------------------------------------------------
// CDNA5 async global->LDS copy (16 bytes per lane), tracked with ASYNCcnt.
// ---------------------------------------------------------------------------
__device__ __forceinline__ void async_copy16(void* lds_ptr, const void* gptr) {
  unsigned loff = (unsigned)(size_t)lds_ptr;
  asm volatile("global_load_async_to_lds_b128 %0, %1, off"
               :: "v"(loff), "v"(gptr) : "memory");
}
__device__ __forceinline__ void wait_async0() {
  asm volatile("s_wait_asynccnt 0x0" ::: "memory");
}

// ---------------------------------------------------------------------------
// WMMA fragment loader. LDS tile layout: [rows][32] bf16.
// 16-bit A 16x32 layout (ISA 7.12.2): lane l -> row = l&15, kb = (l<16)?0:8,
//   VGPR0..3 = K kb..kb+7, VGPR4..7 = K kb+16..kb+23 (two 16B chunks).
// B stored transposed [N][K] uses the identical per-lane addressing.
// ---------------------------------------------------------------------------
__device__ __forceinline__ v16bf load_frag(const __bf16* base, int lane) {
  int r  = lane & 15;
  int kb = (lane >> 4) << 3;           // 0 or 8
  const __bf16* p = base + r * 32;
  union { uint4 u[2]; v16bf v; } t;
  t.u[0] = *reinterpret_cast<const uint4*>(p + kb);
  t.u[1] = *reinterpret_cast<const uint4*>(p + kb + 16);
  return t.v;
}

// ---------------------------------------------------------------------------
// fp32 [K][N] -> bf16 transposed [N][K]  (weight preparation, tiny)
// ---------------------------------------------------------------------------
__global__ __launch_bounds__(256) void cvt_transpose_kernel(const float* __restrict__ in,
                                                            __bf16* __restrict__ out,
                                                            int K, int N) {
  int i = blockIdx.x * blockDim.x + threadIdx.x;
  if (i < K * N) {
    int k = i / N;
    int n = i - k * N;
    out[(size_t)n * K + k] = (__bf16)in[i];
  }
}

// ---------------------------------------------------------------------------
// GEMM1: H = leaky_relu(features @ W1 + b1), bf16 [50000][256].
// Block tile 128(M) x 64(N), K-step 32, 8 waves, LDS double-buffered:
//   per step: issue next A fp32 loads + next B async copy -> WMMA on current
//   buffer -> cvt/ds_store next A -> s_wait_asynccnt -> barrier.
// ---------------------------------------------------------------------------
__global__ __launch_bounds__(256) void gemm1_kernel(const float* __restrict__ A,
                                                    const __bf16* __restrict__ BT, // [256][1280]
                                                    const float* __restrict__ bias,
                                                    __bf16* __restrict__ H, int M) {
  __shared__ __align__(16) __bf16 sA[2][128][32];   // 16 KB
  __shared__ __align__(16) __bf16 sB[2][64][32];    //  8 KB   [N][K]

  const int tid  = threadIdx.x;
  const int wave = tid >> 5;
  const int lane = tid & 31;
  const int n0   = blockIdx.x * 64;    // N fastest -> A band L2-reused across N blocks
  const int m0   = blockIdx.y * 128;

  // Per-thread staging coordinates.
  const int bn  = tid >> 2;            // B chunk row (0..63)
  const int bc8 = (tid & 3) << 3;      // B chunk col offset (bf16 elems)
  int arow[4], acol[4];
  const float* aptr[4];
  for (int i = 0; i < 4; ++i) {
    int j   = tid + i * 256;           // 0..1023 float4 chunks
    arow[i] = j >> 3;                  // 8 float4 per 32-wide row
    acol[i] = (j & 7) << 2;
    int gm  = m0 + arow[i]; if (gm > M - 1) gm = M - 1;   // clamp: branchless OOB
    aptr[i] = &A[(size_t)gm * DIM_FEAT + acol[i]];
  }

  v8f acc[4] = {};

  // Prologue: stage k-step 0 into buffer 0.
  {
    async_copy16(&sB[0][bn][bc8], &BT[(size_t)(n0 + bn) * DIM_FEAT + bc8]);
    for (int i = 0; i < 4; ++i) {
      float4 f = *reinterpret_cast<const float4*>(aptr[i]);
      union { bf16x2 h[2]; uint2 u; } pk;
      pk.h[0] = bf16x2{(__bf16)f.x, (__bf16)f.y};
      pk.h[1] = bf16x2{(__bf16)f.z, (__bf16)f.w};
      *reinterpret_cast<uint2*>(&sA[0][arow[i]][acol[i]]) = pk.u;
    }
    wait_async0();
    __syncthreads();
  }

  const int KSTEPS = DIM_FEAT / 32;    // 40
  for (int step = 0; step < KSTEPS; ++step) {
    const int buf  = step & 1;
    const int nbuf = buf ^ 1;
    const int kn   = (step + 1) * 32;
    const bool has_next = (step + 1) < KSTEPS;

    // 1) Issue next-tile loads (global fp32 A into regs, B via async DMA).
    float4 f[4];
    if (has_next) {
      async_copy16(&sB[nbuf][bn][bc8], &BT[(size_t)(n0 + bn) * DIM_FEAT + kn + bc8]);
      for (int i = 0; i < 4; ++i)
        f[i] = *reinterpret_cast<const float4*>(aptr[i] + kn);
    }

    // 2) Compute on the current buffer.
    v16bf af = load_frag(&sA[buf][wave * 16][0], lane);
    for (int t = 0; t < 4; ++t) {
      v16bf bf = load_frag(&sB[buf][t * 16][0], lane);
      acc[t] = __builtin_amdgcn_wmma_f32_16x16x32_bf16(false, af, false, bf,
                                                       (short)0, acc[t], false, false);
    }

    // 3) Convert + park next A tile behind the math.
    if (has_next) {
      for (int i = 0; i < 4; ++i) {
        union { bf16x2 h[2]; uint2 u; } pk;
        pk.h[0] = bf16x2{(__bf16)f[i].x, (__bf16)f[i].y};
        pk.h[1] = bf16x2{(__bf16)f[i].z, (__bf16)f[i].w};
        *reinterpret_cast<uint2*>(&sA[nbuf][arow[i]][acol[i]]) = pk.u;
      }
    }
    wait_async0();
    __syncthreads();
  }

  // Epilogue: + bias, leaky relu (0.01), store bf16. Uniform fast path:
  // only the last M block can be partial.
  if (m0 + 128 <= M) {
    for (int t = 0; t < 4; ++t) {
      int n = n0 + t * 16 + (lane & 15);
      float bv = bias[n];
      for (int r = 0; r < 8; ++r) {
        int m = m0 + wave * 16 + r + ((lane >> 4) << 3);
        float v = acc[t][r] + bv;
        v = (v >= 0.f) ? v : 0.01f * v;
        H[(size_t)m * H_DIM + n] = (__bf16)v;
      }
    }
  } else {
    for (int t = 0; t < 4; ++t) {
      int n = n0 + t * 16 + (lane & 15);
      float bv = bias[n];
      for (int r = 0; r < 8; ++r) {
        int m = m0 + wave * 16 + r + ((lane >> 4) << 3);
        if (m < M) {
          float v = acc[t][r] + bv;
          v = (v >= 0.f) ? v : 0.01f * v;
          H[(size_t)m * H_DIM + n] = (__bf16)v;
        }
      }
    }
  }
}

// ---------------------------------------------------------------------------
// GEMM2: t = H @ W2 + b2; x[NUM_USER+m][n] = sqrt(|((id^2 + t^2)/2 + 1e-8)|)
// Block tile 128(M) x 64(N==DL), K = 256. Both tiles bf16 -> both staged with
// async global->LDS copies, LDS double-buffered.
// ---------------------------------------------------------------------------
__global__ __launch_bounds__(256) void gemm2_kernel(const __bf16* __restrict__ A,   // [50000][256]
                                                    const __bf16* __restrict__ BT,  // [64][256]
                                                    const float* __restrict__ bias,
                                                    const float* __restrict__ id_embd,
                                                    float* __restrict__ xbuf, int M) {
  __shared__ __align__(16) __bf16 sA[2][128][32];
  __shared__ __align__(16) __bf16 sB[2][64][32];

  const int tid  = threadIdx.x;
  const int wave = tid >> 5;
  const int lane = tid & 31;
  const int m0   = blockIdx.x * 128;

  const int bn  = tid >> 2;
  const int bc8 = (tid & 3) << 3;
  int arow[2], acol[2];
  const __bf16* aptr[2];
  for (int i = 0; i < 2; ++i) {
    int j   = tid + i * 256;           // 0..511 8-elem chunks
    arow[i] = j >> 2;                  // 4 chunks per 32-wide row
    acol[i] = (j & 3) << 3;
    int gm  = m0 + arow[i]; if (gm > M - 1) gm = M - 1;
    aptr[i] = &A[(size_t)gm * H_DIM + acol[i]];
  }

  v8f acc[4] = {};

  auto stage = [&](int buf, int kk) {
    for (int i = 0; i < 2; ++i)
      async_copy16(&sA[buf][arow[i]][acol[i]], aptr[i] + kk);
    async_copy16(&sB[buf][bn][bc8], &BT[(size_t)bn * H_DIM + kk + bc8]);
  };

  stage(0, 0);
  wait_async0();
  __syncthreads();

  const int KSTEPS = H_DIM / 32;       // 8
  for (int step = 0; step < KSTEPS; ++step) {
    const int buf = step & 1;
    if (step + 1 < KSTEPS) stage(buf ^ 1, (step + 1) * 32);

    v16bf af = load_frag(&sA[buf][wave * 16][0], lane);
    for (int t = 0; t < 4; ++t) {
      v16bf bf = load_frag(&sB[buf][t * 16][0], lane);
      acc[t] = __builtin_amdgcn_wmma_f32_16x16x32_bf16(false, af, false, bf,
                                                       (short)0, acc[t], false, false);
    }
    wait_async0();
    __syncthreads();
  }

  if (m0 + 128 <= M) {
    for (int t = 0; t < 4; ++t) {
      int n = t * 16 + (lane & 15);
      float bv = bias[n];
      for (int r = 0; r < 8; ++r) {
        int m = m0 + wave * 16 + r + ((lane >> 4) << 3);
        float v  = acc[t][r] + bv;
        float id = id_embd[(size_t)m * DL + n];
        float o  = sqrtf(fabsf((id * id + v * v) * 0.5f + 1e-8f));
        xbuf[(size_t)(NUM_USER + m) * DL + n] = o;
      }
    }
  } else {
    for (int t = 0; t < 4; ++t) {
      int n = t * 16 + (lane & 15);
      float bv = bias[n];
      for (int r = 0; r < 8; ++r) {
        int m = m0 + wave * 16 + r + ((lane >> 4) << 3);
        if (m < M) {
          float v  = acc[t][r] + bv;
          float id = id_embd[(size_t)m * DL + n];
          float o  = sqrtf(fabsf((id * id + v * v) * 0.5f + 1e-8f));
          xbuf[(size_t)(NUM_USER + m) * DL + n] = o;
        }
      }
    }
  }
}

// ---------------------------------------------------------------------------
// Copy preference into x rows [0, NUM_USER) and into the second output region.
// ---------------------------------------------------------------------------
__global__ __launch_bounds__(256) void copy_pref_kernel(const float* __restrict__ pref,
                                                        float* __restrict__ xbuf,
                                                        float* __restrict__ out_pref, int n) {
  int i = blockIdx.x * blockDim.x + threadIdx.x;
  if (i < n) {
    float v = pref[i];
    xbuf[i]     = v;
    out_pref[i] = v;
  }
}

// ---------------------------------------------------------------------------
// Row L2-normalize (in place -> ego) + seed all_emb and total. One wave per
// row; each lane owns a float2 slice (32 lanes * 2 = 64 dims).
// ---------------------------------------------------------------------------
__global__ __launch_bounds__(256) void normalize_kernel(float* __restrict__ x,
                                                        float* __restrict__ all_emb,
                                                        float* __restrict__ total, int nrows) {
  int w    = (blockIdx.x * blockDim.x + threadIdx.x) >> 5;
  int lane = threadIdx.x & 31;
  if (w >= nrows) return;
  size_t base = (size_t)w * DL + lane * 2;
  float2 v = *reinterpret_cast<float2*>(&x[base]);
  float s = v.x * v.x + v.y * v.y;
  for (int o = 16; o > 0; o >>= 1) s += __shfl_xor(s, o, 32);
  float inv = 1.0f / fmaxf(sqrtf(s), 1e-12f);
  float2 e = make_float2(v.x * inv, v.y * inv);
  *reinterpret_cast<float2*>(&x[base])       = e;
  *reinterpret_cast<float2*>(&all_emb[base]) = e;
  *reinterpret_cast<float2*>(&total[base])   = e;
}

// ---------------------------------------------------------------------------
// SpMM scatter: one wave per edge; lanes cover the 64-dim row in float2
// slices, scatter-add with global_atomic_add_f32 (table is L2-resident).
// ---------------------------------------------------------------------------
__global__ __launch_bounds__(256) void scatter_kernel(const int* __restrict__ erow,
                                                      const int* __restrict__ ecol,
                                                      const float* __restrict__ eval,
                                                      const float* __restrict__ emb,
                                                      float* __restrict__ nxt, int nedges) {
  int e    = (blockIdx.x * blockDim.x + threadIdx.x) >> 5;
  int lane = threadIdx.x & 31;
  if (e >= nedges) return;
  int r = erow[e];
  int c = ecol[e];
  float v = eval[e];
  float2 m = *reinterpret_cast<const float2*>(&emb[(size_t)c * DL + lane * 2]);
  float* dst = &nxt[(size_t)r * DL + lane * 2];
  unsafeAtomicAdd(dst + 0, v * m.x);
  unsafeAtomicAdd(dst + 1, v * m.y);
}

// ---------------------------------------------------------------------------
// Per-row cosine weight vs ego, rescale, accumulate into total, update
// all_emb for the next layer. One wave per row.
// ---------------------------------------------------------------------------
__global__ __launch_bounds__(256) void weight_kernel(const float* __restrict__ nxt,
                                                     const float* __restrict__ ego,
                                                     float* __restrict__ all_emb,
                                                     float* __restrict__ total, int nrows) {
  int w    = (blockIdx.x * blockDim.x + threadIdx.x) >> 5;
  int lane = threadIdx.x & 31;
  if (w >= nrows) return;
  size_t base = (size_t)w * DL + lane * 2;
  float2 a = *reinterpret_cast<const float2*>(&nxt[base]);
  float2 b = *reinterpret_cast<const float2*>(&ego[base]);
  float dot = a.x * b.x + a.y * b.y;
  float na  = a.x * a.x + a.y * a.y;
  float nb  = b.x * b.x + b.y * b.y;
  for (int o = 16; o > 0; o >>= 1) {
    dot += __shfl_xor(dot, o, 32);
    na  += __shfl_xor(na,  o, 32);
    nb  += __shfl_xor(nb,  o, 32);
  }
  float cw = dot / (fmaxf(sqrtf(na), 1e-8f) * fmaxf(sqrtf(nb), 1e-8f));
  float2 nv = make_float2(cw * a.x, cw * a.y);
  *reinterpret_cast<float2*>(&all_emb[base]) = nv;
  float2 t = *reinterpret_cast<float2*>(&total[base]);
  t.x += nv.x; t.y += nv.y;
  *reinterpret_cast<float2*>(&total[base]) = t;
}

// ---------------------------------------------------------------------------
// Host-side orchestration
// ---------------------------------------------------------------------------
extern "C" void kernel_launch(void* const* d_in, const int* in_sizes, int n_in,
                              void* d_out, int out_size, void* d_ws, size_t ws_size,
                              hipStream_t stream) {
  const float* features   = (const float*)d_in[0];
  const float* id_embd    = (const float*)d_in[1];
  const int*   edge_row   = (const int*)d_in[2];
  const int*   edge_col   = (const int*)d_in[3];
  const float* edge_val   = (const float*)d_in[4];
  const float* preference = (const float*)d_in[5];
  const float* W1         = (const float*)d_in[6];
  const float* b1         = (const float*)d_in[7];
  const float* W2         = (const float*)d_in[8];
  const float* b2         = (const float*)d_in[9];

  float* out_total = (float*)d_out;                                // [N_NODES * DL]
  float* out_pref  = (float*)d_out + (size_t)N_NODES * DL;         // [NUM_USER * DL]

  // Workspace carve-up (all sizes multiples of 256 bytes).
  char* ws = (char*)d_ws;
  __bf16* Hbf   = (__bf16*)ws;                           ws += (size_t)NUM_ITEM * H_DIM * 2;   // 25.6 MB
  __bf16* W1T   = (__bf16*)ws;                           ws += (size_t)DIM_FEAT * H_DIM * 2;   // 0.66 MB [256][1280]
  __bf16* W2T   = (__bf16*)ws;                           ws += (size_t)H_DIM * DL * 2;         // 32 KB   [64][256]
  float*  xbuf  = (float*)ws;                            ws += (size_t)N_NODES * DL * 4;       // 38.4 MB (ego)
  float*  emb   = (float*)ws;                            ws += (size_t)N_NODES * DL * 4;       // 38.4 MB (all_emb)
  float*  nxt   = (float*)ws;                            ws += (size_t)N_NODES * DL * 4;       // 38.4 MB (scatter)

  // 1) Weights to bf16, transposed to [N][K] for contiguous B^T tile staging.
  cvt_transpose_kernel<<<(DIM_FEAT * H_DIM + 255) / 256, 256, 0, stream>>>(W1, W1T, DIM_FEAT, H_DIM);
  cvt_transpose_kernel<<<(H_DIM * DL + 255) / 256, 256, 0, stream>>>(W2, W2T, H_DIM, DL);

  // 2) GEMM1 (WMMA bf16): H = leaky_relu(features @ W1 + b1). N is the fast
  //    grid dim so the 4 N-blocks of an M band reuse the A band via L2.
  dim3 g1(H_DIM / 64, (NUM_ITEM + 127) / 128);
  gemm1_kernel<<<g1, 256, 0, stream>>>(features, W1T, b1, Hbf, NUM_ITEM);

  // 3) GEMM2 (WMMA bf16) + fused item-embedding epilogue into x.
  dim3 g2((NUM_ITEM + 127) / 128, 1);
  gemm2_kernel<<<g2, 256, 0, stream>>>(Hbf, W2T, b2, id_embd, xbuf, NUM_ITEM);

  // 4) Preference rows into x and into the second output.
  int npref = NUM_USER * DL;
  copy_pref_kernel<<<(npref + 255) / 256, 256, 0, stream>>>(preference, xbuf, out_pref, npref);

  // 5) ego = l2_normalize(x) in place; seed all_emb and total.
  normalize_kernel<<<((size_t)N_NODES * 32 + 255) / 256, 256, 0, stream>>>(xbuf, emb, out_total, N_NODES);

  // 6) 3 propagation layers: memset -> scatter SpMM -> cosine reweight + accumulate.
  const size_t tbl_bytes = (size_t)N_NODES * DL * sizeof(float);
  for (int l = 0; l < NUM_LAYER; ++l) {
    hipMemsetAsync(nxt, 0, tbl_bytes, stream);
    scatter_kernel<<<((size_t)N_EDGES * 32 + 255) / 256, 256, 0, stream>>>(
        edge_row, edge_col, edge_val, emb, nxt, N_EDGES);
    weight_kernel<<<((size_t)N_NODES * 32 + 255) / 256, 256, 0, stream>>>(
        nxt, xbuf, emb, out_total, N_NODES);
  }
}
